// BitoGuardGNN_18872086298750
// MI455X (gfx1250) — matrix operand order repaired and checked by reading, not compiled, not run on hardware
//
#include <hip/hip_runtime.h>
#include <hip/hip_bf16.h>

// ---------------------------------------------------------------------------
// Types for CDNA5 WMMA (wave32)
// ---------------------------------------------------------------------------
typedef __attribute__((ext_vector_type(16))) __bf16 v16bf;
typedef __attribute__((ext_vector_type(8)))  float  v8f;
typedef __attribute__((ext_vector_type(4)))  unsigned int u32x4;
typedef int async_v4i __attribute__((vector_size(16)));   // builtin's pointee type

union FragU {
    u32x4 q[2];
    v16bf v;
};

static constexpr int NUn = 50000;
static constexpr int NWn = 50000;
static constexpr int En  = 100000;

// ---------------------------------------------------------------------------
// CDNA5 async global->LDS path (ASYNCcnt), with synchronous fallback
// ---------------------------------------------------------------------------
#if defined(__HIP_DEVICE_COMPILE__) && \
    __has_builtin(__builtin_amdgcn_global_load_async_to_lds_b128) && \
    __has_builtin(__builtin_amdgcn_s_wait_asynccnt)
#define HAS_ASYNC_LDS 1
#else
#define HAS_ASYNC_LDS 0
#endif

__device__ __forceinline__ void cp16_g2l(const unsigned short* g, unsigned short* l) {
#if HAS_ASYNC_LDS
    __builtin_amdgcn_global_load_async_to_lds_b128(
        (__attribute__((address_space(1))) async_v4i*)g,
        (__attribute__((address_space(3))) async_v4i*)l, 0, 0);
#else
    *(u32x4*)l = *(const u32x4*)g;
#endif
}
__device__ __forceinline__ void wait_async_lds() {
#if HAS_ASYNC_LDS
    __builtin_amdgcn_s_wait_asynccnt(0);
#endif
}

// ---------------------------------------------------------------------------
// device helpers
// ---------------------------------------------------------------------------
__device__ __forceinline__ unsigned short f2bf(float f) {
    unsigned u = __float_as_uint(f);
    unsigned r = u + 0x7FFFu + ((u >> 16) & 1u);   // round-to-nearest-even
    return (unsigned short)(r >> 16);
}
// order-preserving float->uint key (monotone), key(x)=0 is below all reals
__device__ __forceinline__ unsigned fkey(float f) {
    unsigned b = __float_as_uint(f);
    return (b & 0x80000000u) ? ~b : (b | 0x80000000u);
}
__device__ __forceinline__ float keyf(unsigned k) {
    unsigned b = (k & 0x80000000u) ? (k & 0x7FFFFFFFu) : ~k;
    return __uint_as_float(b);
}

// ---------------------------------------------------------------------------
// input projection: out[n,j] = b[j] + sum_k x[n,k]*W[j,k]   (j = 0..127)
// also emits bf16 copy for downstream WMMA GEMMs
// ---------------------------------------------------------------------------
__global__ __launch_bounds__(128) void proj_kernel(
    const float* __restrict__ x, const float* __restrict__ W,
    const float* __restrict__ b, int K_, float* __restrict__ out,
    unsigned short* __restrict__ obf) {
    __shared__ float xs[64];
    const int n = blockIdx.x;
    const int j = threadIdx.x;
    if (j < K_) xs[j] = x[(size_t)n * K_ + j];
    __syncthreads();
    float a = b[j];
    for (int k = 0; k < K_; ++k) a += xs[k] * W[j * K_ + k];
    out[(size_t)n * 128 + j] = a;
    obf[(size_t)n * 128 + j] = f2bf(a);
}

// ---------------------------------------------------------------------------
// weight convert+transpose: W [N_,K_] f32 row-major -> Bt [K_,N_] bf16
// ---------------------------------------------------------------------------
__global__ void wcvt_kernel(const float* __restrict__ W, int N_, int K_,
                            unsigned short* __restrict__ Bt) {
    int idx = blockIdx.x * blockDim.x + threadIdx.x;
    if (idx >= N_ * K_) return;
    int n = idx / K_;
    int k = idx - n * K_;
    Bt[(size_t)k * N_ + n] = f2bf(W[idx]);
}

// ---------------------------------------------------------------------------
// bf16 WMMA GEMM: C[M,N] = A[M,K]_bf16 @ Bt[K,N]_bf16 + bias[N]   (f32 out)
// block = 256 threads (8 waves). Block tile 128x128, wave tile 32x64,
// k-step 32. Double-buffered LDS fed by GLOBAL_LOAD_ASYNC_TO_LDS_B128:
// tile k+1 streams into LDS while the matrix pipe runs tile k.
// ---------------------------------------------------------------------------
__global__ __launch_bounds__(256) void gemm_bf16_kernel(
    const unsigned short* __restrict__ A, const unsigned short* __restrict__ Bt,
    const float* __restrict__ bias, float* __restrict__ C,
    int M, int N, int K) {
    __shared__ __align__(16) unsigned short As[2][128 * 32];
    __shared__ __align__(16) unsigned short Bs[2][32 * 128];

    const int tid  = threadIdx.x;
    const int lane = tid & 31;
    const int wave = tid >> 5;
    const int wm   = wave & 3;   // 0..3 -> rows wm*32
    const int wn   = wave >> 2;  // 0..1 -> cols wn*64
    const int m0   = blockIdx.x * 128;
    const int n0   = blockIdx.y * 128;

    // pre-zero LDS rows past the M boundary once (async copies skip them)
    if (m0 + 128 > M) {
        for (int c = tid; c < 512; c += 256) {
            int row = c >> 2, col = (c & 3) << 3;
            if (m0 + row >= M) {
                u32x4 z = {};
                *(u32x4*)(&As[0][row * 32 + col]) = z;
                *(u32x4*)(&As[1][row * 32 + col]) = z;
            }
        }
    }

    auto stage = [&](int b_, int kk) {
        // A tile 128x32 bf16: 512 chunks of 16B
        for (int c = tid; c < 512; c += 256) {
            int row = c >> 2, col = (c & 3) << 3;
            int gr  = m0 + row;
            if (gr < M)
                cp16_g2l(A + (size_t)gr * K + kk + col, &As[b_][row * 32 + col]);
        }
        // B tile 32x128 bf16
        for (int c = tid; c < 512; c += 256) {
            int row = c >> 4, col = (c & 15) << 3;
            cp16_g2l(Bt + (size_t)(kk + row) * N + n0 + col, &Bs[b_][row * 128 + col]);
        }
    };

    v8f acc[2][4] = {};
    const int am = lane & 15;   // A fragment row within 16
    const int ah = lane >> 4;   // A fragment half-lane group

    stage(0, 0);
    int buf = 0;
    for (int kk = 0; kk < K; kk += 32) {
        wait_async_lds();            // this wave's async copies landed
        __syncthreads();             // all waves' copies of tile[buf] landed
        const int cur = buf;
        if (kk + 32 < K) {           // prefetch next tile into other buffer
            buf ^= 1;
            stage(buf, kk + 32);
        }

        // ---- fragments (layouts per CDNA5 ISA 7.12.2)
        FragU a[2], b[4];
        for (int mi = 0; mi < 2; ++mi) {
            const unsigned short* p =
                &As[cur][(wm * 32 + mi * 16 + am) * 32 + ah * 8];
            a[mi].q[0] = *(const u32x4*)p;        // K = g0 + 0..7
            a[mi].q[1] = *(const u32x4*)(p + 16); // K = g0 + 16..23
        }
        for (int ni = 0; ni < 4; ++ni) {
            const unsigned short* p =
                &Bs[cur][(ah * 8) * 128 + wn * 64 + ni * 16 + am];
            for (int i = 0; i < 4; ++i) {         // col N = am, K = g0 + 0..7 / g0 + 16..23
                b[ni].q[0][i] = (unsigned)p[(2 * i) * 128] | ((unsigned)p[(2 * i + 1) * 128] << 16);
                b[ni].q[1][i] = (unsigned)p[(16 + 2 * i) * 128] | ((unsigned)p[(16 + 2 * i + 1) * 128] << 16);
            }
        }
        for (int mi = 0; mi < 2; ++mi)
            for (int ni = 0; ni < 4; ++ni)
                acc[mi][ni] = __builtin_amdgcn_wmma_f32_16x16x32_bf16(
                    false, a[mi].v, false, b[ni].v, (short)0, acc[mi][ni],
                    false, false);
        __syncthreads();             // readers done before buf is overwritten
    }

    // ---- epilogue: D layout -> row = r + 8*(lane>>4), col = lane&15
    const int cl = lane & 15;
    const int ch = lane >> 4;
    for (int mi = 0; mi < 2; ++mi) {
        for (int ni = 0; ni < 4; ++ni) {
            int coln = n0 + wn * 64 + ni * 16 + cl;
            float bv = bias ? bias[coln] : 0.0f;
            for (int r = 0; r < 8; ++r) {
                int row = m0 + wm * 32 + mi * 16 + ch * 8 + r;
                if (row < M)
                    C[(size_t)row * N + coln] = acc[mi][ni][r] + bv;
            }
        }
    }
}

// ---------------------------------------------------------------------------
// edge pass 1: alpha[e,h] = sum_c att[h,c]*leaky_relu(xl[src]+xr[dst]+ea*We)
//              + segment max into mkey[dst,h] (ordered-uint atomicMax)
// one wave per edge; C == 128 for all convs here.
// ---------------------------------------------------------------------------
__global__ __launch_bounds__(256) void edge_alpha_kernel(
    const float* __restrict__ xl, const float* __restrict__ xr,
    const float* __restrict__ ea, const float* __restrict__ We,
    const float* __restrict__ att, const int* __restrict__ ei,
    int E_, int H, int HC, float* __restrict__ alpha,
    unsigned* __restrict__ mkey) {
    int e = blockIdx.x * 8 + (threadIdx.x >> 5);
    if (e >= E_) return;
    const int lane = threadIdx.x & 31;
    const int src = ei[e];
    const int dst = ei[E_ + e];
    const float eav = ea[e];

    float acc[4] = {0.f, 0.f, 0.f, 0.f};
    const int iters = HC >> 5;
    for (int i = 0; i < iters; ++i) {
        int j = (i << 5) + lane;
        float z = xl[(size_t)src * HC + j] + xr[(size_t)dst * HC + j] + eav * We[j];
        z = z > 0.f ? z : 0.2f * z;
        acc[i >> 2] += z * att[j];   // C=128 -> head = i/4
    }
    for (int h = 0; h < H; ++h) {
        float r = acc[h];
        r += __shfl_xor(r, 16);
        r += __shfl_xor(r, 8);
        r += __shfl_xor(r, 4);
        r += __shfl_xor(r, 2);
        r += __shfl_xor(r, 1);
        if (lane == 0) {
            alpha[(size_t)e * H + h] = r;
            atomicMax(&mkey[(size_t)dst * H + h], fkey(r));
        }
    }
}

// ---------------------------------------------------------------------------
// edge pass 2: p = exp(alpha - m[dst]); segment sum s[dst,h]
// ---------------------------------------------------------------------------
__global__ void edge_softmax_kernel(const int* __restrict__ ei, int E_, int H,
                                    const unsigned* __restrict__ mkey,
                                    float* __restrict__ alpha_p,
                                    float* __restrict__ s) {
    int idx = blockIdx.x * blockDim.x + threadIdx.x;
    if (idx >= E_ * H) return;
    int e = idx / H;
    int h = idx - e * H;
    int dst = ei[E_ + e];
    float m = keyf(mkey[(size_t)dst * H + h]);
    float p = __expf(alpha_p[idx] - m);
    alpha_p[idx] = p;
    atomicAdd(&s[(size_t)dst * H + h], p);
}

// ---------------------------------------------------------------------------
// edge pass 3: acc[dst, h*128+c] += xl[src, h*128+c] * p/max(s,1e-16)
// ---------------------------------------------------------------------------
__global__ __launch_bounds__(256) void edge_aggregate_kernel(
    const float* __restrict__ xl, const int* __restrict__ ei,
    const float* __restrict__ p, const float* __restrict__ s,
    int E_, int H, int HC, float* __restrict__ acc) {
    int e = blockIdx.x * 8 + (threadIdx.x >> 5);
    if (e >= E_) return;
    const int lane = threadIdx.x & 31;
    const int src = ei[e];
    const int dst = ei[E_ + e];
    float w[4];
    for (int h = 0; h < H; ++h) {
        float sv = s[(size_t)dst * H + h];
        w[h] = p[(size_t)e * H + h] / fmaxf(sv, 1e-16f);
    }
    const int iters = HC >> 5;
    for (int i = 0; i < iters; ++i) {
        int j = (i << 5) + lane;
        atomicAdd(&acc[(size_t)dst * HC + j],
                  xl[(size_t)src * HC + j] * w[i >> 2]);
    }
}

// ---------------------------------------------------------------------------
// finalize: v = relu(acc + b1 [+ b2]); write f32 in place, optional bf16 copy
// ---------------------------------------------------------------------------
__global__ void finalize_kernel(float* __restrict__ acc,
                                const float* __restrict__ b1,
                                const float* __restrict__ b2,
                                int total, int HC,
                                unsigned short* __restrict__ obf) {
    int idx = blockIdx.x * blockDim.x + threadIdx.x;
    if (idx >= total) return;
    int j = idx % HC;
    float v = acc[idx] + b1[j] + (b2 ? b2[j] : 0.0f);
    v = v > 0.f ? v : 0.f;
    acc[idx] = v;
    if (obf) obf[idx] = f2bf(v);
}

// ---------------------------------------------------------------------------
// output: gate = sigmoid(cs*gW + gb); y = (u2*gate) @ oW.T + ob
// ---------------------------------------------------------------------------
__global__ __launch_bounds__(128) void out_kernel(
    const float* __restrict__ u2, const float* __restrict__ cs,
    const float* __restrict__ gW, const float* __restrict__ gb,
    const float* __restrict__ oW, const float* __restrict__ ob,
    float* __restrict__ y) {
    __shared__ float g[128];
    const int n = blockIdx.x;
    const int j = threadIdx.x;
    float x = cs[n] * gW[j] + gb[j];
    float gate = 1.0f / (1.0f + __expf(-x));
    g[j] = u2[(size_t)n * 128 + j] * gate;
    __syncthreads();
    if (j < 64) {
        float a = ob[j];
        for (int k = 0; k < 128; ++k) a += g[k] * oW[j * 128 + k];
        y[(size_t)n * 64 + j] = a;
    }
}

// ---------------------------------------------------------------------------
// host orchestration
// ---------------------------------------------------------------------------
extern "C" void kernel_launch(void* const* d_in, const int* in_sizes, int n_in,
                              void* d_out, int out_size, void* d_ws, size_t ws_size,
                              hipStream_t stream) {
    const float* x_user = (const float*)d_in[0];
    const float* x_wall = (const float*)d_in[1];
    const float* ea_st  = (const float*)d_in[2];
    const float* ea_rf  = (const float*)d_in[3];
    const float* ea_tw  = (const float*)d_in[4];
    const float* cate   = (const float*)d_in[5];
    const float* pWu = (const float*)d_in[6];
    const float* pbu = (const float*)d_in[7];
    const float* pWw = (const float*)d_in[8];
    const float* pbw = (const float*)d_in[9];
    const float* l1_Wl   = (const float*)d_in[10];
    const float* l1_bl   = (const float*)d_in[11];
    const float* l1_Wr   = (const float*)d_in[12];
    const float* l1_br   = (const float*)d_in[13];
    const float* l1_We   = (const float*)d_in[14];
    const float* l1_att  = (const float*)d_in[15];
    const float* l1_bias = (const float*)d_in[16];
    const float* l2_Wl   = (const float*)d_in[17];
    const float* l2_bl   = (const float*)d_in[18];
    const float* l2_Wr   = (const float*)d_in[19];
    const float* l2_br   = (const float*)d_in[20];
    const float* l2_We   = (const float*)d_in[21];
    const float* l2_att  = (const float*)d_in[22];
    const float* l2_bias = (const float*)d_in[23];
    const float* gW = (const float*)d_in[24];
    const float* gb = (const float*)d_in[25];
    const float* oW = (const float*)d_in[26];
    const float* ob = (const float*)d_in[27];
    const int* ei_st = (const int*)d_in[28];
    const int* ei_rf = (const int*)d_in[29];
    const int* ei_tw = (const int*)d_in[30];

    // ---- workspace carving (256B aligned)
    char* ws = (char*)d_ws;
    size_t off = 0;
    auto alloc = [&](size_t bytes) -> void* {
        void* p = ws + off;
        off = (off + bytes + 255) & ~(size_t)255;
        return p;
    };
    float*          hu     = (float*)alloc((size_t)NUn * 128 * 4);
    float*          hw     = (float*)alloc((size_t)NWn * 128 * 4);
    unsigned short* hu_bf  = (unsigned short*)alloc((size_t)NUn * 128 * 2);
    unsigned short* hw_bf  = (unsigned short*)alloc((size_t)NWn * 128 * 2);
    float*          bufA   = (float*)alloc((size_t)50000 * 512 * 4);  // xl
    float*          bufB   = (float*)alloc((size_t)50000 * 512 * 4);  // xr
    float*          accW1  = (float*)alloc((size_t)NWn * 512 * 4);
    float*          accU1  = (float*)alloc((size_t)NUn * 512 * 4);
    unsigned short* w1_bf  = (unsigned short*)alloc((size_t)NWn * 512 * 2);
    unsigned short* u1_bf  = (unsigned short*)alloc((size_t)NUn * 512 * 2);
    float*          accU2  = (float*)alloc((size_t)NUn * 128 * 4);
    float*          alphaP = (float*)alloc((size_t)En * 4 * 4);
    unsigned*       mkey   = (unsigned*)alloc((size_t)50000 * 4 * 4);
    float*          sbuf   = (float*)alloc((size_t)50000 * 4 * 4);
    unsigned short* BtL    = (unsigned short*)alloc((size_t)512 * 128 * 2);
    unsigned short* BtR    = (unsigned short*)alloc((size_t)512 * 128 * 2);
    float*          xl2    = bufA;  // reuse (layer-1 xl/xr dead by then)
    float*          xr2    = bufB;

    auto gemm = [&](const unsigned short* A, const unsigned short* Bt,
                    const float* bias, float* C, int M, int N, int K) {
        dim3 g((M + 127) / 128, N / 128);
        gemm_bf16_kernel<<<g, 256, 0, stream>>>(A, Bt, bias, C, M, N, K);
    };
    auto wcvt = [&](const float* W, int N_, int K_, unsigned short* Bt) {
        int tot = N_ * K_;
        wcvt_kernel<<<(tot + 255) / 256, 256, 0, stream>>>(W, N_, K_, Bt);
    };
    auto run_conv = [&](const float* xl, const float* xr, const int* ei,
                        const float* ea, const float* We, const float* att,
                        int nd, int H, int HC, float* acc) {
        (void)hipMemsetAsync(mkey, 0, (size_t)nd * H * 4, stream);
        (void)hipMemsetAsync(sbuf, 0, (size_t)nd * H * 4, stream);
        int eb = (En + 7) / 8;
        edge_alpha_kernel<<<eb, 256, 0, stream>>>(xl, xr, ea, We, att, ei,
                                                  En, H, HC, alphaP, mkey);
        int t2 = En * H;
        edge_softmax_kernel<<<(t2 + 255) / 256, 256, 0, stream>>>(ei, En, H,
                                                                  mkey, alphaP, sbuf);
        edge_aggregate_kernel<<<eb, 256, 0, stream>>>(xl, ei, alphaP, sbuf,
                                                      En, H, HC, acc);
    };

    // ---- input projections
    proj_kernel<<<NUn, 128, 0, stream>>>(x_user, pWu, pbu, 50, hu, hu_bf);
    proj_kernel<<<NWn, 128, 0, stream>>>(x_wall, pWw, pbw, 10, hw, hw_bf);

    // ================= layer 1 : st (wallet <- user), heads=4 =================
    wcvt(l1_Wl + 0 * 65536, 512, 128, BtL);
    wcvt(l1_Wr + 0 * 65536, 512, 128, BtR);
    gemm(hu_bf, BtL, l1_bl + 0, bufA, NUn, 512, 128);   // xl = lin_l(hu)
    gemm(hw_bf, BtR, l1_br + 0, bufB, NWn, 512, 128);   // xr = lin_r(hw)
    (void)hipMemsetAsync(accW1, 0, (size_t)NWn * 512 * 4, stream);
    run_conv(bufA, bufB, ei_st, ea_st, l1_We + 0, l1_att + 0, NWn, 4, 512, accW1);
    finalize_kernel<<<(NWn * 512 + 255) / 256, 256, 0, stream>>>(
        accW1, l1_bias + 0, nullptr, NWn * 512, 512, w1_bf);   // w1 = relu(.)

    // ================= layer 1 : rf + tw (user), heads=4 =================
    (void)hipMemsetAsync(accU1, 0, (size_t)NUn * 512 * 4, stream);
    wcvt(l1_Wl + 1 * 65536, 512, 128, BtL);
    wcvt(l1_Wr + 1 * 65536, 512, 128, BtR);
    gemm(hw_bf, BtL, l1_bl + 512, bufA, NWn, 512, 128); // xl = lin_l(hw)
    gemm(hu_bf, BtR, l1_br + 512, bufB, NUn, 512, 128); // xr = lin_r(hu)
    run_conv(bufA, bufB, ei_rf, ea_rf, l1_We + 512, l1_att + 512, NUn, 4, 512, accU1);

    wcvt(l1_Wl + 2 * 65536, 512, 128, BtL);
    wcvt(l1_Wr + 2 * 65536, 512, 128, BtR);
    gemm(hu_bf, BtL, l1_bl + 1024, bufA, NUn, 512, 128);
    gemm(hu_bf, BtR, l1_br + 1024, bufB, NUn, 512, 128);
    run_conv(bufA, bufB, ei_tw, ea_tw, l1_We + 1024, l1_att + 1024, NUn, 4, 512, accU1);
    finalize_kernel<<<(NUn * 512 + 255) / 256, 256, 0, stream>>>(
        accU1, l1_bias + 512, l1_bias + 1024, NUn * 512, 512, u1_bf); // u1

    // ================= layer 2 : rf + tw (user), heads=1 =================
    (void)hipMemsetAsync(accU2, 0, (size_t)NUn * 128 * 4, stream);
    wcvt(l2_Wl + 1 * 65536, 128, 512, BtL);
    wcvt(l2_Wr + 1 * 65536, 128, 512, BtR);
    gemm(w1_bf, BtL, l2_bl + 128, xl2, NWn, 128, 512);  // xl = lin_l(w1)
    gemm(u1_bf, BtR, l2_br + 128, xr2, NUn, 128, 512);  // xr = lin_r(u1)
    run_conv(xl2, xr2, ei_rf, ea_rf, l2_We + 128, l2_att + 128, NUn, 1, 128, accU2);

    wcvt(l2_Wl + 2 * 65536, 128, 512, BtL);
    wcvt(l2_Wr + 2 * 65536, 128, 512, BtR);
    gemm(u1_bf, BtL, l2_bl + 256, xl2, NUn, 128, 512);
    gemm(u1_bf, BtR, l2_br + 256, xr2, NUn, 128, 512);
    run_conv(xl2, xr2, ei_tw, ea_tw, l2_We + 256, l2_att + 256, NUn, 1, 128, accU2);
    finalize_kernel<<<(NUn * 128 + 255) / 256, 256, 0, stream>>>(
        accU2, l2_bias + 128, l2_bias + 256, NUn * 128, 128, nullptr);  // u2

    // ================= gated output =================
    out_kernel<<<NUn, 128, 0, stream>>>(accU2, cate, gW, gb, oW, ob,
                                        (float*)d_out);
}
